// GQACrossAttention_63763084476555
// MI455X (gfx1250) — compile-verified, matrix operand order checked by reading, not run on hardware
//
#include <hip/hip_runtime.h>
#include <hip/hip_bf16.h>
#include <cstdint>

typedef __attribute__((ext_vector_type(16))) __bf16        v16bf;
typedef __attribute__((ext_vector_type(8)))  float         v8f;
typedef __attribute__((ext_vector_type(4)))  unsigned int  u32x4;
typedef __attribute__((ext_vector_type(8)))  int           i32x8;
typedef __attribute__((ext_vector_type(4)))  int           i32x4;

#define B_   4
#define S_   1024
#define T_   2048
#define HID_ 1024
#define NH_  16
#define NKV_ 4
#define HD_  64
#define REP_ (NH_ / NKV_)

// ---------------------------------------------------------------------------
// Tensor Data Mover: 2-D bf16 tile (tile_d0 x tile_d1 elements) from global
// (row stride = stride_elems) into LDS at lds_off, with an 8-element (16B)
// LDS pad appended after every 64-element (32-dword) row:
//   pad_interval code 4 -> 32 dwords, pad_amount code 3 -> 4 dwords.
// Descriptor layout per CDNA5 ISA 8.3/8.4. Issued by one wave, tracked by
// TENSORcnt; completion is in-order, so double-buffering can wait on "<=1".
// ---------------------------------------------------------------------------
__device__ __forceinline__ void tdm_load_tile_bf16(
    unsigned lds_off, const void* gptr,
    unsigned tile_d0, unsigned tile_d1, unsigned stride_elems)
{
    const unsigned long long ga = (unsigned long long)(uintptr_t)gptr;
    u32x4 g0;
    g0[0] = 1u;                                   // count=1 (valid user D#)
    g0[1] = lds_off;                              // lds_addr (bytes)
    g0[2] = (unsigned)ga;                         // global_addr[31:0]
    g0[3] = (unsigned)(ga >> 32) | (2u << 30);    // global_addr[56:32] | type=2
    i32x8 g1;
    g1[0] = (1 << 16)                             // data_size = 2 bytes (bf16)
          | (1 << 20)                             // pad_enable
          | (4 << 22)                             // pad_interval = 32 dwords
          | (3 << 25);                            // pad_amount   = 4 dwords
    g1[1] = (int)(tile_d0 << 16);                 // tensor_dim0[15:0]
    g1[2] = (int)(tile_d1 << 16);                 // tensor_dim1[15:0]
    g1[3] = (int)(tile_d0 << 16);                 // tile_dim0
    g1[4] = (int)tile_d1;                         // tile_dim1 (tile_dim2=0)
    g1[5] = (int)stride_elems;                    // tensor_dim0_stride[31:0]
    g1[6] = 0;
    g1[7] = 0;
    const i32x4 z4 = {0, 0, 0, 0};
#if __clang_major__ >= 23
    const i32x8 z8 = {0, 0, 0, 0, 0, 0, 0, 0};
    __builtin_amdgcn_tensor_load_to_lds(g0, g1, z4, z4, z8, 0);
#else
    __builtin_amdgcn_tensor_load_to_lds(g0, g1, z4, z4, 0);
#endif
}

// ---------------------------------------------------------------------------
// Max-reduce across each 16-lane row group using full-rate VALU DPP permutes
// (no LDS traffic): quad xor1, quad xor2, row_half_mirror, row_mirror.
// ---------------------------------------------------------------------------
__device__ __forceinline__ float rowmax16(float x)
{
    int v = __builtin_bit_cast(int, x);
    int p = __builtin_amdgcn_update_dpp(v, v, 0xB1, 0xF, 0xF, true);   // [1,0,3,2]
    x = fmaxf(x, __builtin_bit_cast(float, p));
    v = __builtin_bit_cast(int, x);
    p = __builtin_amdgcn_update_dpp(v, v, 0x4E, 0xF, 0xF, true);       // [2,3,0,1]
    x = fmaxf(x, __builtin_bit_cast(float, p));
    v = __builtin_bit_cast(int, x);
    p = __builtin_amdgcn_update_dpp(v, v, 0x141, 0xF, 0xF, true);      // half_mirror
    x = fmaxf(x, __builtin_bit_cast(float, p));
    v = __builtin_bit_cast(int, x);
    p = __builtin_amdgcn_update_dpp(v, v, 0x140, 0xF, 0xF, true);      // mirror
    x = fmaxf(x, __builtin_bit_cast(float, p));
    return x;
}

// ---------------------------------------------------------------------------
// C[M,N] = A[M,K] * W[N,K]^T   (bf16 WMMA compute, f32 accumulate)
// InT:  float or __bf16 (A operand); W is always f32 (harness weights).
// Block: 256 threads = 8 waves; tile 64(M) x 64(N); K-step 64.
// Fragment layout per CDNA5 ISA 7.12.2:
//   lane L: row L%16, kbase=8*(L/16), chunks [kbase,+8) & [kbase+16,+24)
// ---------------------------------------------------------------------------
template <typename InT, typename OutT>
__global__ __launch_bounds__(256) void gemm_nt_wmma(
    const InT* __restrict__ A, const float* __restrict__ W,
    OutT* __restrict__ C, int M, int N, int K)
{
    __shared__ __bf16 As[64][72];   // 64 rows x 64 cols (+8 pad)
    __shared__ __bf16 Ws[64][72];

    const int tid  = threadIdx.x;
    const int wave = tid >> 5;
    const int lane = tid & 31;
    const int half = lane >> 4;
    const int l16  = lane & 15;

    const int m0 = blockIdx.x * 64;
    const int n0 = blockIdx.y * 64;

    const int waveM = (wave & 3) * 16;   // 0,16,32,48
    const int waveN = (wave >> 2) * 32;  // 0,32

    v8f acc0 = {};
    v8f acc1 = {};

    const int ldRow = tid >> 2;          // 0..63
    const int ldCol = (tid & 3) * 16;    // 0,16,32,48

    for (int k0 = 0; k0 < K; k0 += 64) {
        const InT*   ap = A + (size_t)(m0 + ldRow) * K + k0 + ldCol;
        const float* wp = W + (size_t)(n0 + ldRow) * K + k0 + ldCol;
#pragma unroll
        for (int j = 0; j < 16; ++j) {
            As[ldRow][ldCol + j] = (__bf16)ap[j];
            Ws[ldRow][ldCol + j] = (__bf16)wp[j];
        }
        __syncthreads();

        const int kb = half * 8;
#pragma unroll
        for (int s = 0; s < 2; ++s) {
            const int c0 = s * 32 + kb;
            v16bf af, bf0, bf1;
#pragma unroll
            for (int e = 0; e < 8; ++e) {
                af[e]      = As[waveM + l16][c0 + e];
                af[e + 8]  = As[waveM + l16][c0 + 16 + e];
                bf0[e]     = Ws[waveN + l16][c0 + e];
                bf0[e + 8] = Ws[waveN + l16][c0 + 16 + e];
                bf1[e]     = Ws[waveN + 16 + l16][c0 + e];
                bf1[e + 8] = Ws[waveN + 16 + l16][c0 + 16 + e];
            }
            acc0 = __builtin_amdgcn_wmma_f32_16x16x32_bf16(false, af, false, bf0,
                                                           (short)0, acc0, false, false);
            acc1 = __builtin_amdgcn_wmma_f32_16x16x32_bf16(false, af, false, bf1,
                                                           (short)0, acc1, false, false);
        }
        __syncthreads();
    }

    // C-fragment: lane L holds rows 8*(L/16)+r (r=0..7), col L%16
#pragma unroll
    for (int r = 0; r < 8; ++r) {
        const size_t row = (size_t)(m0 + waveM + half * 8 + r);
        C[row * N + n0 + waveN + l16]      = (OutT)acc0[r];
        C[row * N + n0 + waveN + 16 + l16] = (OutT)acc1[r];
    }
}

// ---------------------------------------------------------------------------
// Flash attention (GQA: query head h reads KV head g = h/REP_).
// Grid: (S/64, B*NH). Block: 128 threads = 4 waves; wave w owns query rows
// [16w,16w+16) of the 64-row block, full HD=64.
// Per 32-wide T tile:
//   K tile staged by the Tensor Data Mover, double-buffered: tile i+1 is
//     issued before compute of tile i; s_wait_tensorcnt(1) retires tile i.
//   V tile staged cooperatively transposed (pure bf16 copy);
//   scores = 4 WMMAs; row-max via DPP; P round-trips per-wave LDS
//   (C-layout -> A-layout); row-sum = 1 WMMA against an all-ones B fragment
//   (every lane receives the row sum, no shuffles); O += P*V = 4 WMMAs.
// ---------------------------------------------------------------------------
__global__ __launch_bounds__(128) void gqa_flash_attn_wmma(
    const __bf16* __restrict__ Q,    // [B*S, NH*HD]
    const __bf16* __restrict__ Kb,   // [B*T, NKV*HD]
    const __bf16* __restrict__ Vb,   // [B*T, NKV*HD]
    __bf16* __restrict__ O)          // [B*S, NH*HD]
{
    __shared__ __bf16 Ks[2][32][72];    // TDM double buffer: [t][d] + row pad
    __shared__ __bf16 Vt[64][40];       // [d][t] (transposed)
    __shared__ __bf16 Ps[4][16][40];    // per-wave P round-trip buffer

    const int tid  = threadIdx.x;
    const int wave = tid >> 5;
    const int lane = tid & 31;
    const int half = lane >> 4;
    const int l16  = lane & 15;

    const int head = blockIdx.y;            // b*NH + h
    const int b = head / NH_;
    const int h = head % NH_;
    const int g = h / REP_;

    const int qRow0  = b * S_ + blockIdx.x * 64 + wave * 16;
    const int qCol0  = h * HD_;
    const int kvCol0 = g * HD_;

    const unsigned ksOff[2] = { (unsigned)(uintptr_t)(&Ks[0][0][0]),
                                (unsigned)(uintptr_t)(&Ks[1][0][0]) };
    const __bf16* kBase = Kb + (size_t)(b * T_) * (NKV_ * HD_) + kvCol0;

    // preload Q fragments (A-layout), softmax scale 1/sqrt(64) folded in
    v16bf qf[2];
    {
        const __bf16* qp = Q + (size_t)(qRow0 + l16) * (NH_ * HD_) + qCol0;
#pragma unroll
        for (int s = 0; s < 2; ++s) {
            const int kb = s * 32 + half * 8;
#pragma unroll
            for (int e = 0; e < 8; ++e) {
                qf[s][e]     = (__bf16)((float)qp[kb + e]      * 0.125f);
                qf[s][e + 8] = (__bf16)((float)qp[kb + 16 + e] * 0.125f);
            }
        }
    }

    // all-ones B fragment: (P x ones) puts the P row-sum in every lane
    v16bf ones;
#pragma unroll
    for (int e = 0; e < 16; ++e) ones[e] = (__bf16)1.0f;

    v8f o[4] = {{}, {}, {}, {}};
    float mrow[8], lrow[8];
#pragma unroll
    for (int r = 0; r < 8; ++r) { mrow[r] = -1e30f; lrow[r] = 0.0f; }

    const int ldr = tid >> 2;          // 0..31  (V tile row)
    const int ldc = (tid & 3) * 16;    // 0,16,32,48

    const int NIT = T_ / 32;

    // prime the TDM pipeline with tile 0
    if (wave == 0)
        tdm_load_tile_bf16(ksOff[0], kBase, 64u, 32u, (unsigned)(NKV_ * HD_));

    for (int it = 0; it < NIT; ++it) {
        const int t0 = it * 32;

        // issue DMA for the next K tile into the other buffer
        if (wave == 0 && it + 1 < NIT)
            tdm_load_tile_bf16(ksOff[(it + 1) & 1],
                               kBase + (size_t)(t0 + 32) * (NKV_ * HD_),
                               64u, 32u, (unsigned)(NKV_ * HD_));

        // stage V tile transposed (pure bf16 copy)
        {
            const __bf16* vp = Vb + (size_t)(b * T_ + t0 + ldr) * (NKV_ * HD_)
                                  + kvCol0 + ldc;
#pragma unroll
            for (int j = 0; j < 16; ++j)
                Vt[ldc + j][ldr] = vp[j];
        }

        if (wave == 0) {
            if (it + 1 < NIT) __builtin_amdgcn_s_wait_tensorcnt(1); // tile it done
            else              __builtin_amdgcn_s_wait_tensorcnt(0);
        }
        __syncthreads();

        const __bf16 (*Kc)[72] = Ks[it & 1];

        // ---- scores S = Q * K^T : 16x32 per wave ----
        v8f s0 = {}, s1 = {};
        const int kb = half * 8;
#pragma unroll
        for (int s = 0; s < 2; ++s) {
            v16bf b0, b1;
#pragma unroll
            for (int e = 0; e < 8; ++e) {
                b0[e]     = Kc[l16][s * 32 + kb + e];
                b0[e + 8] = Kc[l16][s * 32 + kb + 16 + e];
                b1[e]     = Kc[16 + l16][s * 32 + kb + e];
                b1[e + 8] = Kc[16 + l16][s * 32 + kb + 16 + e];
            }
            s0 = __builtin_amdgcn_wmma_f32_16x16x32_bf16(false, qf[s], false, b0,
                                                         (short)0, s0, false, false);
            s1 = __builtin_amdgcn_wmma_f32_16x16x32_bf16(false, qf[s], false, b1,
                                                         (short)0, s1, false, false);
        }

        // ---- online softmax: row-max via DPP, exponentials, rescale O ----
        float p0[8], p1[8], alpha[8];
#pragma unroll
        for (int r = 0; r < 8; ++r) {
            const float a = s0[r];
            const float c = s1[r];
            const float mx = rowmax16(fmaxf(a, c));
            const float nm = fmaxf(mrow[r], mx);
            p0[r]    = __expf(a - nm);
            p1[r]    = __expf(c - nm);
            alpha[r] = __expf(mrow[r] - nm);
            mrow[r]  = nm;
            o[0][r] *= alpha[r]; o[1][r] *= alpha[r];
            o[2][r] *= alpha[r]; o[3][r] *= alpha[r];
        }

        // ---- P: C-layout -> A-layout via per-wave LDS (in-wave, dscnt) ----
#pragma unroll
        for (int r = 0; r < 8; ++r) {
            Ps[wave][half * 8 + r][l16]      = (__bf16)p0[r];
            Ps[wave][half * 8 + r][16 + l16] = (__bf16)p1[r];
        }
        asm volatile("s_wait_dscnt 0" ::: "memory");
        v16bf pf;
#pragma unroll
        for (int e = 0; e < 8; ++e) {
            pf[e]     = Ps[wave][l16][kb + e];
            pf[e + 8] = Ps[wave][l16][kb + 16 + e];
        }

        // ---- row-sum of P via WMMA against all-ones (lands in every lane) ----
        v8f srow = {};
        srow = __builtin_amdgcn_wmma_f32_16x16x32_bf16(false, pf, false, ones,
                                                       (short)0, srow, false, false);
#pragma unroll
        for (int r = 0; r < 8; ++r)
            lrow[r] = lrow[r] * alpha[r] + srow[r];

        // ---- O += P * V : four 16-wide d-subtiles, K=32 ----
#pragma unroll
        for (int t = 0; t < 4; ++t) {
            v16bf vf;
#pragma unroll
            for (int e = 0; e < 8; ++e) {
                vf[e]     = Vt[t * 16 + l16][kb + e];
                vf[e + 8] = Vt[t * 16 + l16][kb + 16 + e];
            }
            o[t] = __builtin_amdgcn_wmma_f32_16x16x32_bf16(false, pf, false, vf,
                                                           (short)0, o[t], false, false);
        }
        __syncthreads();
    }

    // ---- epilogue: O / l, stored bf16 for the Wo GEMM ----
#pragma unroll
    for (int r = 0; r < 8; ++r) {
        const float inv = 1.0f / lrow[r];
        const size_t row = (size_t)(qRow0 + half * 8 + r);
        __bf16* op = O + row * (NH_ * HD_) + qCol0;
#pragma unroll
        for (int t = 0; t < 4; ++t)
            op[t * 16 + l16] = (__bf16)(o[t][r] * inv);
    }
}

// ---------------------------------------------------------------------------
extern "C" void kernel_launch(void* const* d_in, const int* in_sizes, int n_in,
                              void* d_out, int out_size, void* d_ws, size_t ws_size,
                              hipStream_t stream)
{
    const float* x   = (const float*)d_in[0];   // [B,S,HID]
    const float* ctx = (const float*)d_in[1];   // [B,T,HID]
    const float* Wq  = (const float*)d_in[2];   // [NH*HD, HID]
    const float* Wk  = (const float*)d_in[3];   // [NKV*HD, HID]
    const float* Wv  = (const float*)d_in[4];   // [NKV*HD, HID]
    const float* Wo  = (const float*)d_in[5];   // [HID, NH*HD]
    float* out = (float*)d_out;

    // bf16 workspace: q(8MB) | k(4MB) | v(4MB) | attn_out(8MB)  (~24MB)
    __bf16* q  = (__bf16*)d_ws;
    __bf16* k  = q + (size_t)(B_ * S_) * (NH_ * HD_);
    __bf16* v  = k + (size_t)(B_ * T_) * (NKV_ * HD_);
    __bf16* ao = v + (size_t)(B_ * T_) * (NKV_ * HD_);

    const dim3 blk(256);
    // q = x @ Wq^T            [4096,1024] x [1024,1024]^T -> bf16
    gemm_nt_wmma<float, __bf16><<<dim3((B_ * S_) / 64, (NH_ * HD_) / 64), blk, 0, stream>>>(
        x, Wq, q, B_ * S_, NH_ * HD_, HID_);
    // k = ctx @ Wk^T          [8192,1024] x [256,1024]^T -> bf16
    gemm_nt_wmma<float, __bf16><<<dim3((B_ * T_) / 64, (NKV_ * HD_) / 64), blk, 0, stream>>>(
        ctx, Wk, k, B_ * T_, NKV_ * HD_, HID_);
    // v = ctx @ Wv^T -> bf16
    gemm_nt_wmma<float, __bf16><<<dim3((B_ * T_) / 64, (NKV_ * HD_) / 64), blk, 0, stream>>>(
        ctx, Wv, v, B_ * T_, NKV_ * HD_, HID_);
    // attention (flash, double-buffered TDM K tiles, no score materialization)
    gqa_flash_attn_wmma<<<dim3(S_ / 64, B_ * NH_), dim3(128), 0, stream>>>(q, k, v, ao);
    // out = attn_out @ Wo^T   [4096,1024](bf16) x [1024,1024]^T -> f32
    gemm_nt_wmma<__bf16, float><<<dim3((B_ * S_) / 64, HID_ / 64), blk, 0, stream>>>(
        ao, Wo, out, B_ * S_, HID_, NH_ * HD_);
}
